// Head_85109071937988
// MI455X (gfx1250) — matrix operand order repaired and checked by reading, not compiled, not run on hardware
//
#include <hip/hip_runtime.h>

// MI455X / gfx1250: wave32, WMMA f16 16x16x32 -> f32 accumulate,
// async global->LDS staging (ASYNCcnt), software-pipelined operand loads.
typedef __attribute__((ext_vector_type(16))) _Float16 v16h;
typedef __attribute__((ext_vector_type(8)))  _Float16 v8h;
typedef __attribute__((ext_vector_type(8)))  float    v8f;
typedef __attribute__((ext_vector_type(4)))  float    v4f;

#define BDIM 512
#define TDIM 256
#define EDIM 384
#define HDIM 64

// Issue one async 16-byte global->LDS copy (tracked by ASYNCcnt).
__device__ __forceinline__ void async_cp16(unsigned lds_addr, const void* gptr) {
  asm volatile("global_load_async_to_lds_b128 %0, %1, off"
               :: "v"(lds_addr), "v"((unsigned long long)(uintptr_t)gptr)
               : "memory");
}
__device__ __forceinline__ void wait_async0() {
  asm volatile("s_wait_asynccnt 0" ::: "memory");
}
// Low 32 bits of a generic pointer to __shared__ == LDS byte address.
__device__ __forceinline__ unsigned lds_addr_of(const void* p) {
  return (unsigned)(uintptr_t)p;
}

// ---------------------------------------------------------------------------
// Kernel 0: repack Wq|Wk|Wv (f32 [E][H]) into f16 tiles in exact WMMA B-operand
// layout (32x16 k x n): lane<16 holds k={0..7,16..23}, lane>=16 holds k={8..15,24..31}, n=lane%16.
// One tile = 32 lanes x 16 halves.  tileIndex = (mat*4 + nt)*12 + kc.
// ---------------------------------------------------------------------------
__global__ __launch_bounds__(32) void pack_w(const float* __restrict__ Wq,
                                             const float* __restrict__ Wk,
                                             const float* __restrict__ Wv,
                                             _Float16* __restrict__ wpk) {
  const int tile = blockIdx.x;                 // 0..143
  const int mat  = tile / 48;
  const int nt   = (tile % 48) / 12;
  const int kc   = tile % 12;
  const float* W = (mat == 0) ? Wq : (mat == 1) ? Wk : Wv;
  const int lane = threadIdx.x;
  const int n    = lane & 15;
  const int kb   = kc * 32 + ((lane < 16) ? 0 : 8);
  _Float16* dst  = wpk + (size_t)tile * 512 + lane * 16;
#pragma unroll
  for (int i = 0; i < 16; ++i)
    dst[i] = (_Float16)W[(size_t)(kb + i + ((i < 8) ? 0 : 8)) * HDIM + nt * 16 + n];
}

// ---------------------------------------------------------------------------
// Kernel 1: QKV projection.  Wave = 16-row strip of x; 12 live accumulators
// (Q/K/V x 4 n-tiles).  Packed W (12 KB per 32-wide E chunk) double-buffered
// in LDS via async copies; inside the chunk the B-tile DS loads are software-
// pipelined depth-2 so each v_wmma overlaps the next tile's ds_load_b128 pair.
// ---------------------------------------------------------------------------
__global__ __launch_bounds__(256) void qkv_kernel(const float* __restrict__ x,
                                                  const _Float16* __restrict__ wpk,
                                                  _Float16* __restrict__ qw,
                                                  _Float16* __restrict__ kw,
                                                  _Float16* __restrict__ vt) {
  __shared__ __align__(16) _Float16 wbuf[2][12 * 512];   // 2 x 12 KB

  const int tid   = threadIdx.x;
  const int lane  = tid & 31;
  const int wave  = tid >> 5;
  const int strip = blockIdx.x * 8 + wave;      // 0..8191
  const int row0  = strip * 16;                 // flat row = b*T + t
  const int mrow  = row0 + (lane & 15);
  const int loff  = (lane < 16) ? 0 : 8;

  // cooperative async stage of one 12 KB W chunk (768 x 16B, 3 per thread)
  auto stage = [&](int buf, int kc) {
    const char* src = (const char*)(wpk + (size_t)kc * 12 * 512);
    const unsigned dst = lds_addr_of(&wbuf[buf][0]);
#pragma unroll
    for (int j = 0; j < 3; ++j) {
      const int chunk = tid + j * 256;
      async_cp16(dst + chunk * 16, src + chunk * 16);
    }
  };

  stage(0, 0);
  v8f acc[12] = {};

  for (int kc = 0; kc < 12; ++kc) {
    wait_async0();        // chunk kc resident in LDS (this wave's copies)
    __syncthreads();      // ... and everyone else's; prev buffer fully consumed
    if (kc + 1 < 12) stage((kc + 1) & 1, kc + 1);

    // A tile 16x32 from x (f32->f16): contiguous runs loff+{0..7},{16..23}.
    const float* xr = x + (size_t)mrow * EDIM + kc * 32 + loff;
    v4f f0 = *(const v4f*)(xr);
    v4f f1 = *(const v4f*)(xr + 4);
    v4f f2 = *(const v4f*)(xr + 16);
    v4f f3 = *(const v4f*)(xr + 20);
    v16h a;
#pragma unroll
    for (int i = 0; i < 4; ++i) {
      a[i]      = (_Float16)f0[i];
      a[4 + i]  = (_Float16)f1[i];
      a[8 + i]  = (_Float16)f2[i];
      a[12 + i] = (_Float16)f3[i];
    }

    // depth-2 pipelined B-tile loads feeding 12 WMMAs
    const _Float16* wb = &wbuf[kc & 1][0];
    v16h bcur = *(const v16h*)(wb + lane * 16);
#pragma unroll
    for (int t = 0; t < 12; ++t) {
      v16h bnext = bcur;
      if (t + 1 < 12)
        bnext = *(const v16h*)(wb + (t + 1) * 512 + lane * 16);
      acc[t] = __builtin_amdgcn_wmma_f32_16x16x32_f16(false, a, false, bcur,
                                                      (short)0, acc[t], false, false);
      bcur = bnext;
    }
  }

  // D layout: VGPR r -> row r (lanes 0-15) / r+8 (lanes 16-31), col = lane%16.
  const int nlo  = lane & 15;
  const int radd = (lane < 16) ? 0 : 8;
#pragma unroll
  for (int t = 0; t < 12; ++t) {
    const int mat = t >> 2, nt = t & 3;
#pragma unroll
    for (int r = 0; r < 8; ++r) {
      const int g = row0 + r + radd;
      const int h = nt * 16 + nlo;
      const _Float16 v = (_Float16)acc[t][r];
      if (mat == 0)      qw[(size_t)g * HDIM + h] = v;
      else if (mat == 1) kw[(size_t)g * HDIM + h] = v;
      else {
        const int b = g >> 8, tt = g & 255;
        vt[((size_t)b * HDIM + h) * TDIM + tt] = v;   // transposed store
      }
    }
  }
}

// ---------------------------------------------------------------------------
// Kernel 2: causal attention, online softmax.  Wave = 16 query rows.
// V_b (32 KB) async-staged once per block into LDS with rows padded to 264
// halves (132 DW stride -> B-operand reads cover all 64 banks once).
// K_b stays in global (L2-resident).  All 4 B tiles of each WMMA group are
// fetched up-front so the waits descend instead of stalling per tile.
// ---------------------------------------------------------------------------
__global__ __launch_bounds__(256) void attn_kernel(const _Float16* __restrict__ qw,
                                                   const _Float16* __restrict__ kw,
                                                   const _Float16* __restrict__ vt,
                                                   float* __restrict__ out) {
  __shared__ __align__(16) _Float16 vlds[HDIM][264];   // 33792 B, padded rows
  __shared__ __align__(16) _Float16 plds[8][16][32];   // per-wave P scratch

  const int tid  = threadIdx.x;
  const int lane = tid & 31;
  const int wave = tid >> 5;
  const int b    = blockIdx.x >> 1;
  const int q0   = (blockIdx.x & 1) * 128 + wave * 16;
  const int nlo  = lane & 15;
  const int loff = (lane < 16) ? 0 : 16;   // A-operand k offset
  const int koff = (lane < 16) ? 0 : 16;   // B-operand k offset
  const int radd = (lane < 16) ? 0 : 8;    // D row offset

  const _Float16* kb = kw + (size_t)b * TDIM * HDIM;
  const _Float16* vb = vt + (size_t)b * HDIM * TDIM;

  // async-stage V_b: 2048 x 16B chunks, 8 per thread
#pragma unroll
  for (int j = 0; j < 8; ++j) {
    const int chunk = tid + j * 256;          // 0..2047
    const int row = chunk >> 5, c16 = chunk & 31;
    async_cp16(lds_addr_of(&vlds[row][c16 * 8]), vb + (size_t)row * TDIM + c16 * 8);
  }

  // Q strip as A operand (registers), overlapped with the async V copy.
  v16h qa[2];
#pragma unroll
  for (int hc = 0; hc < 2; ++hc) {
    const _Float16* qp = qw + ((size_t)b * TDIM + q0 + nlo) * HDIM + hc * 32 + loff;
    v8h lo = *(const v8h*)(qp);
    v8h hi = *(const v8h*)(qp + 8);
#pragma unroll
    for (int i = 0; i < 8; ++i) { qa[hc][i] = lo[i]; qa[hc][8 + i] = hi[i]; }
  }

  wait_async0();
  __syncthreads();

  v8f o[4] = {};
  float mrow[8], lrow[8];
#pragma unroll
  for (int r = 0; r < 8; ++r) { mrow[r] = -__builtin_inff(); lrow[r] = 0.f; }

  const int nchunks = (q0 + 15) / 32 + 1;    // causal: skip fully-masked chunks

  for (int c = 0; c < nchunks; ++c) {
    const int tb = c * 32;

    // S = Q·K^T : fetch all 4 K B-tiles first, then 4 WMMAs.
    v16h kbm[4];
#pragma unroll
    for (int tt = 0; tt < 2; ++tt)
#pragma unroll
      for (int hc = 0; hc < 2; ++hc)
        kbm[tt * 2 + hc] =
            *(const v16h*)(kb + (size_t)(tb + tt * 16 + nlo) * HDIM + hc * 32 + koff);

    v8f s[2] = {};
#pragma unroll
    for (int tt = 0; tt < 2; ++tt)
#pragma unroll
      for (int hc = 0; hc < 2; ++hc)
        s[tt] = __builtin_amdgcn_wmma_f32_16x16x32_f16(false, qa[hc], false,
                                                       kbm[tt * 2 + hc],
                                                       (short)0, s[tt], false, false);

    // scale + causal mask + per-lane tile max
    float tmax[8];
#pragma unroll
    for (int r = 0; r < 8; ++r) {
      const int q  = q0 + r + radd;
      const int t0 = tb + nlo, t1 = tb + 16 + nlo;
      const float v0 = (t0 <= q) ? s[0][r] * 0.125f : -__builtin_inff();
      const float v1 = (t1 <= q) ? s[1][r] * 0.125f : -__builtin_inff();
      s[0][r] = v0; s[1][r] = v1;
      tmax[r] = fmaxf(v0, v1);
    }
#pragma unroll
    for (int r = 0; r < 8; ++r) {
#pragma unroll
      for (int d = 1; d < 16; d <<= 1)
        tmax[r] = fmaxf(tmax[r], __shfl_xor(tmax[r], d, 16));
    }

    float alpha[8];
#pragma unroll
    for (int r = 0; r < 8; ++r) {
      const float mn = fmaxf(mrow[r], tmax[r]);
      alpha[r] = __expf(mrow[r] - mn);       // exp(-inf)=0 on first chunk
      mrow[r]  = mn;
    }

    // P = exp(S - m): row sums + store P to per-wave LDS (row-major 16x32)
    float rsum[8];
#pragma unroll
    for (int r = 0; r < 8; ++r) {
      const float p0 = __expf(s[0][r] - mrow[r]);
      const float p1 = __expf(s[1][r] - mrow[r]);
      rsum[r] = p0 + p1;
      const int q = r + radd;
      plds[wave][q][nlo]      = (_Float16)p0;
      plds[wave][q][16 + nlo] = (_Float16)p1;
    }
#pragma unroll
    for (int r = 0; r < 8; ++r) {
#pragma unroll
      for (int d = 1; d < 16; d <<= 1)
        rsum[r] += __shfl_xor(rsum[r], d, 16);
      lrow[r] = lrow[r] * alpha[r] + rsum[r];
    }

    // rescale running output
#pragma unroll
    for (int ht = 0; ht < 4; ++ht)
#pragma unroll
      for (int r = 0; r < 8; ++r) o[ht][r] *= alpha[r];

    asm volatile("s_wait_dscnt 0" ::: "memory");  // P store -> P reload (cross-lane)

    // reload P in A-operand layout, and prefetch all 4 V B-tiles from LDS
    v16h pa;
    {
      const _Float16* pp = &plds[wave][nlo][loff];
      v8h lo = *(const v8h*)(pp);
      v8h hi = *(const v8h*)(pp + 8);
#pragma unroll
      for (int i = 0; i < 8; ++i) { pa[i] = lo[i]; pa[8 + i] = hi[i]; }
    }
    v16h vbm[4];
#pragma unroll
    for (int ht = 0; ht < 4; ++ht) {
      const _Float16* vp = &vlds[ht * 16 + nlo][tb + koff];
      v8h lo = *(const v8h*)(vp);
      v8h hi = *(const v8h*)(vp + 8);
#pragma unroll
      for (int i = 0; i < 8; ++i) { vbm[ht][i] = lo[i]; vbm[ht][8 + i] = hi[i]; }
    }

    // O += P·V
#pragma unroll
    for (int ht = 0; ht < 4; ++ht)
      o[ht] = __builtin_amdgcn_wmma_f32_16x16x32_f16(false, pa, false, vbm[ht],
                                                     (short)0, o[ht], false, false);
  }

  // epilogue: normalize and store f32 output [B][T][H]
#pragma unroll
  for (int ht = 0; ht < 4; ++ht) {
#pragma unroll
    for (int r = 0; r < 8; ++r) {
      const int q = q0 + r + radd;
      const int h = ht * 16 + nlo;
      out[((size_t)b * TDIM + q) * HDIM + h] = o[ht][r] / lrow[r];
    }
  }
}

// ---------------------------------------------------------------------------
// Launch.  Workspace (f16): [0,147456) packed W; then qw, kw, vt (16 MB each).
// ---------------------------------------------------------------------------
extern "C" void kernel_launch(void* const* d_in, const int* in_sizes, int n_in,
                              void* d_out, int out_size, void* d_ws, size_t ws_size,
                              hipStream_t stream) {
  const float* x  = (const float*)d_in[0];
  const float* Wq = (const float*)d_in[1];
  const float* Wk = (const float*)d_in[2];
  const float* Wv = (const float*)d_in[3];
  float* out = (float*)d_out;

  char* ws = (char*)d_ws;
  const size_t QKV_BYTES = (size_t)BDIM * TDIM * HDIM * sizeof(_Float16); // 16 MB
  _Float16* wpk = (_Float16*)(ws);
  _Float16* qw  = (_Float16*)(ws + (256u << 10));
  _Float16* kw  = (_Float16*)(ws + (256u << 10) + QKV_BYTES);
  _Float16* vt  = (_Float16*)(ws + (256u << 10) + 2 * QKV_BYTES);

  pack_w<<<dim3(144), dim3(32), 0, stream>>>(Wq, Wk, Wv, wpk);
  qkv_kernel<<<dim3(BDIM * TDIM / 128), dim3(256), 0, stream>>>(x, wpk, qw, kw, vt);
  attn_kernel<<<dim3(BDIM * 2), dim3(256), 0, stream>>>(qw, kw, vt, out);
}
